// Model_GBM_16819091931179
// MI455X (gfx1250) — matrix operand order repaired and checked by reading, not compiled, not run on hardware
//
#include <hip/hip_runtime.h>
#include <hip/hip_bf16.h>

// ---------------------------------------------------------------------------
// GBM basket BSDE, d=100, H=512, B=4096, N=50.
// y-recursion is linear -> all 50 steps' MLPs are independent:
//   one batched MLP with M = 204800 rows on v_wmma_f32_16x16x32_f16.
// The Brownian correlation dW = sqrt(dt)*L*Wi is ALSO done as a WMMA GEMM:
//   C[(b,j), k] = sum_m WiT[(b,j), m] * (sqrt(dt)*L)[k, m].
// ---------------------------------------------------------------------------

typedef __attribute__((ext_vector_type(16))) _Float16 v16h;
typedef __attribute__((ext_vector_type(8)))  float    v8f;

#define BATCH   4096
#define NSTEP   50
#define HDIM    512
#define DPAD    112   // 100 padded to multiple of 16
#define KU      128   // 101 padded to multiple of 32

// ---- WMMA fragment helpers (wave32, 16x16x32 f16) -------------------------

// A fragment: 16x32 (MxK), row-major source with leading dim ld (elements).
__device__ __forceinline__ v16h loadA(const _Float16* p, int ld) {
    const int l    = threadIdx.x & 31;
    const int row  = l & 15;
    const int half = l >> 4;
    union { v16h v; unsigned int u[8]; } f;
    const unsigned int* r = (const unsigned int*)(p + row * ld);
#pragma unroll
    for (int v = 0; v < 8; ++v) {
        const int k = ((v & 3) * 2) + half * 8 + ((v >> 2) * 16);
        f.u[v] = r[k >> 1];
    }
    return f.v;
}

// B fragment: 32x16 (KxN). Source is weight row-major [N][K] (B = W^T).
__device__ __forceinline__ v16h loadB(const _Float16* p, int ld) {
    const int l = threadIdx.x & 31;
    const int n = l & 15;
    const int g = l >> 4;
    union { v16h v; unsigned int u[8]; } f;
    const unsigned int* r = (const unsigned int*)(p + n * ld);
#pragma unroll
    for (int v = 0; v < 8; ++v) {
        const int k = ((v & 3) * 2) + g * 8 + ((v >> 2) * 16);
        f.u[v] = r[k >> 1];
    }
    return f.v;
}

__device__ __forceinline__ v8f wmma_f16(v16h a, v16h b, v8f c) {
    return __builtin_amdgcn_wmma_f32_16x16x32_f16(
        false, a, false, b, (short)0, c, false, false);
}

// ---- Kernel 1: weights + L -> f16 with zero padding ------------------------
// W1h [512][128] (cols 101..127=0), W2h [512][512], W3h [112][512]
// (rows 100..111=0), Lh [112][128] = sqrt(dt)*L padded.
__global__ void cvt_weights(const float* __restrict__ W1,
                            const float* __restrict__ W2,
                            const float* __restrict__ W3,
                            const float* __restrict__ L,
                            _Float16* __restrict__ W1h,
                            _Float16* __restrict__ W2h,
                            _Float16* __restrict__ W3h,
                            _Float16* __restrict__ Lh) {
    const int idx = blockIdx.x * 256 + threadIdx.x;
    if (idx < 512 * KU) {
        const int n = idx >> 7, k = idx & (KU - 1);
        W1h[idx] = (_Float16)((k < 101) ? W1[n * 101 + k] : 0.0f);
        return;
    }
    const int i2 = idx - 512 * KU;
    if (i2 >= 0 && i2 < 512 * 512) {
        W2h[i2] = (_Float16)W2[i2];
        return;
    }
    const int i3 = idx - 512 * KU - 512 * 512;
    if (i3 >= 0 && i3 < DPAD * 512) {
        const int n = i3 >> 9;
        W3h[i3] = (_Float16)((n < 100) ? W3[i3] : 0.0f);
        return;
    }
    const int i4 = idx - 512 * KU - 512 * 512 - DPAD * 512;
    if (i4 >= 0 && i4 < DPAD * KU) {
        const int k = i4 >> 7, m = i4 & (KU - 1);
        const float sdt = 0.141421356237f;  // sqrt(dt), dt = 0.02
        Lh[i4] = (_Float16)((k < 100 && m < 100) ? sdt * L[k * 100 + m] : 0.0f);
    }
}

// ---- Kernel 2a: transpose W_indep[b][m][j] -> WiT[(b*50+j)][128] f16 -------
__global__ __launch_bounds__(256) void transpose_wi(const float* __restrict__ Wi,
                                                    _Float16* __restrict__ WiT) {
    __shared__ float w[100 * NSTEP];
    const int b = blockIdx.x;
    const float* src = Wi + (size_t)b * (100 * NSTEP);
    for (int idx = threadIdx.x; idx < 100 * NSTEP; idx += 256) w[idx] = src[idx];
    __syncthreads();
    _Float16* dst = WiT + (size_t)b * NSTEP * KU;
    for (int o = threadIdx.x; o < NSTEP * KU; o += 256) {
        const int j = o >> 7, m = o & (KU - 1);
        dst[o] = (_Float16)((m < 100) ? w[m * NSTEP + j] : 0.0f);
    }
}

// ---- Kernel 2b: dWpad[(b*50+i)][k] = WiT @ Lh^T (M=204800,N=112,K=128) ----
__global__ __launch_bounds__(512, 2)
void corr_gemm(const _Float16* __restrict__ WiT,
               const _Float16* __restrict__ Lh,
               float* __restrict__ dWpad) {
    __shared__ _Float16 Alds[64 * KU];
    const int tid  = threadIdx.x;
    const int wave = tid >> 5;
    const int lane = tid & 31;
    const size_t r0 = (size_t)blockIdx.x * 64;

    {
        const unsigned int* src = (const unsigned int*)(WiT + r0 * KU);
        unsigned int* dst = (unsigned int*)Alds;
        for (int idx = tid; idx < 64 * KU / 2; idx += 512) dst[idx] = src[idx];
    }
    __syncthreads();

    for (int t = wave; t < 28; t += 16) {
        const int mt = t / 7, nt = t - mt * 7;
        const int n0 = nt * 16;
        v8f acc = {0.f, 0.f, 0.f, 0.f, 0.f, 0.f, 0.f, 0.f};
#pragma unroll
        for (int kk = 0; kk < 4; ++kk) {
            v16h a = loadA(Alds + mt * 16 * KU + kk * 32, KU);
            v16h b = loadB(Lh + (size_t)n0 * KU + kk * 32, KU);
            acc = wmma_f16(a, b, acc);
        }
        const int n  = n0 + (lane & 15);
        const int mr = mt * 16 + (lane >> 4) * 8;
#pragma unroll
        for (int r = 0; r < 8; ++r)
            dWpad[(r0 + mr + r) * DPAD + n] = acc[r];
    }
}

// ---- Kernel 3: GBM paths -> f16 MLP inputs + final x -----------------------
// xin[i][b][0..99] = x_i (f16), [100] = t_i, [101..127] = 0.
// dWpad is (b,i)-major: dWpad[(b*50+i)*112 + k].
__global__ __launch_bounds__(128) void paths(const float* __restrict__ dWpad,
                                             _Float16* __restrict__ xin,
                                             float* __restrict__ xout) {
    const int b = blockIdx.x, d = threadIdx.x;
    const float drift = (0.05f - 0.5f * 0.2f * 0.2f) * 0.02f;  // 6e-4
    if (d < 100) {
        float s = 0.0f;
        for (int i = 0; i < NSTEP; ++i) {
            const float x = 100.0f * __expf(drift * (float)i + 0.2f * s);
            xin[((size_t)i * BATCH + b) * KU + d] = (_Float16)x;
            s += dWpad[((size_t)b * NSTEP + i) * DPAD + d];
        }
        xout[(size_t)b * 100 + d] = 100.0f * __expf(drift * (float)NSTEP + 0.2f * s);
    } else if (d == 100) {
        for (int i = 0; i < NSTEP; ++i)
            xin[((size_t)i * BATCH + b) * KU + d] = (_Float16)(0.02f * (float)i);
    } else {
        for (int i = 0; i < NSTEP; ++i)
            xin[((size_t)i * BATCH + b) * KU + d] = (_Float16)0.0f;
    }
}

// ---- Kernel 4: fused 3-layer MLP + z.w epilogue ---------------------------
// One WG = 64 batch rows x 1 step. 16 waves. LDS: U 16KB + H1 64KB + H2 64KB.
__global__ __launch_bounds__(512, 1)
void mlp_fused(const _Float16* __restrict__ xin,
               const _Float16* __restrict__ W1h, const float* __restrict__ b1,
               const _Float16* __restrict__ W2h, const float* __restrict__ b2,
               const _Float16* __restrict__ W3h, const float* __restrict__ b3,
               const float* __restrict__ dWpad, float* __restrict__ zw) {
    __shared__ _Float16 Ulds[64 * KU];
    __shared__ _Float16 H1[64 * HDIM];
    __shared__ _Float16 H2[64 * HDIM];

    const int tid   = threadIdx.x;
    const int wave  = tid >> 5;
    const int lane  = tid & 31;
    const int istep = blockIdx.y;
    const int b0    = blockIdx.x * 64;

    // Warm L2/L1 for the big weight matrix (global_prefetch_b8).
    __builtin_prefetch(W2h + (size_t)tid * HDIM, 0, 1);

    // Stage U block (contiguous 64x128 f16 rows).
    {
        const unsigned int* src =
            (const unsigned int*)(xin + ((size_t)istep * BATCH + b0) * KU);
        unsigned int* dst = (unsigned int*)Ulds;
        for (int idx = tid; idx < 64 * KU / 2; idx += 512) dst[idx] = src[idx];
    }
    __syncthreads();

    // ---- Layer 1: H1 = relu(U @ W1^T + b1), M=64 N=512 K=128 ----
    for (int mt = 0; mt < 4; ++mt) {
        const _Float16* abase = Ulds + mt * 16 * KU;
        v16h a0 = loadA(abase +  0, KU);
        v16h a1 = loadA(abase + 32, KU);
        v16h a2 = loadA(abase + 64, KU);
        v16h a3 = loadA(abase + 96, KU);
        for (int nn = 0; nn < 2; ++nn) {
            const int n0 = (wave * 2 + nn) * 16;
            v8f acc = {0.f, 0.f, 0.f, 0.f, 0.f, 0.f, 0.f, 0.f};
            const _Float16* wb = W1h + (size_t)n0 * KU;
            acc = wmma_f16(a0, loadB(wb +  0, KU), acc);
            acc = wmma_f16(a1, loadB(wb + 32, KU), acc);
            acc = wmma_f16(a2, loadB(wb + 64, KU), acc);
            acc = wmma_f16(a3, loadB(wb + 96, KU), acc);
            const int n    = n0 + (lane & 15);
            const float bv = b1[n];
            const int mr   = mt * 16 + (lane >> 4) * 8;
#pragma unroll
            for (int r = 0; r < 8; ++r)
                H1[(mr + r) * HDIM + n] = (_Float16)fmaxf(acc[r] + bv, 0.0f);
        }
    }
    __syncthreads();

    // ---- Layer 2: H2 = relu(H1 @ W2^T + b2), M=64 N=512 K=512 ----
    for (int mt = 0; mt < 4; ++mt) {
        for (int nn = 0; nn < 2; ++nn) {
            const int n0 = (wave * 2 + nn) * 16;
            v8f acc = {0.f, 0.f, 0.f, 0.f, 0.f, 0.f, 0.f, 0.f};
            for (int kk = 0; kk < 16; ++kk) {
                v16h a = loadA(H1 + mt * 16 * HDIM + kk * 32, HDIM);
                v16h b = loadB(W2h + (size_t)n0 * HDIM + kk * 32, HDIM);
                acc = wmma_f16(a, b, acc);
            }
            const int n    = n0 + (lane & 15);
            const float bv = b2[n];
            const int mr   = mt * 16 + (lane >> 4) * 8;
#pragma unroll
            for (int r = 0; r < 8; ++r)
                H2[(mr + r) * HDIM + n] = (_Float16)fmaxf(acc[r] + bv, 0.0f);
        }
    }
    __syncthreads();

    // ---- Layer 3: Z = H2 @ W3^T + b3, M=64 N=112 K=512 (no relu) ----
    float* zbuf = (float*)H1;  // H1 is dead; 64*112 f32 = 28KB < 64KB
    for (int t = wave; t < 28; t += 16) {
        const int mt = t / 7, nt = t - mt * 7;
        const int n0 = nt * 16;
        v8f acc = {0.f, 0.f, 0.f, 0.f, 0.f, 0.f, 0.f, 0.f};
        for (int kk = 0; kk < 16; ++kk) {
            v16h a = loadA(H2 + mt * 16 * HDIM + kk * 32, HDIM);
            v16h b = loadB(W3h + (size_t)n0 * HDIM + kk * 32, HDIM);
            acc = wmma_f16(a, b, acc);
        }
        const int n    = n0 + (lane & 15);
        const float bv = (n < 100) ? b3[n] : 0.0f;
        const int mr   = mt * 16 + (lane >> 4) * 8;
#pragma unroll
        for (int r = 0; r < 8; ++r)
            zbuf[(mr + r) * DPAD + n] = acc[r] + bv;
    }
    __syncthreads();

    // ---- Epilogue: zw[row] = sum_n Z[row,n] * w[row,n] (pads are 0) ----
    {
        float* psum = zbuf + 64 * DPAD;  // +512 f32, still inside H1 region
        const int row = tid >> 3, c = tid & 7;
        const float* wv =
            dWpad + ((size_t)(b0 + row) * NSTEP + istep) * DPAD + c * 14;
        const float* zr = zbuf + row * DPAD + c * 14;
        float s = 0.0f;
#pragma unroll
        for (int n = 0; n < 14; ++n) s += zr[n] * wv[n];
        psum[tid] = s;
        __syncthreads();
        if (tid < 64) {
            float acc = 0.0f;
#pragma unroll
            for (int c2 = 0; c2 < 8; ++c2) acc += psum[tid * 8 + c2];
            zw[(size_t)istep * BATCH + b0 + tid] = acc;
        }
    }
}

// ---- Kernel 5: fold zw into the linear y-recursion ------------------------
__global__ void finalize(const float* __restrict__ zw,
                         const float* __restrict__ y0,
                         float* __restrict__ yout) {
    const int b = blockIdx.x * 256 + threadIdx.x;
    if (b < BATCH) {
        float y = y0[0];
        for (int i = 0; i < NSTEP; ++i)
            y = y * 1.001f + zw[(size_t)i * BATCH + b];  // 1 + r*dt
        yout[b] = y;
    }
}

// ---------------------------------------------------------------------------
extern "C" void kernel_launch(void* const* d_in, const int* in_sizes, int n_in,
                              void* d_out, int out_size, void* d_ws, size_t ws_size,
                              hipStream_t stream) {
    const float* W_indep = (const float*)d_in[0];
    const float* W1 = (const float*)d_in[1];
    const float* b1 = (const float*)d_in[2];
    const float* W2 = (const float*)d_in[3];
    const float* b2 = (const float*)d_in[4];
    const float* W3 = (const float*)d_in[5];
    const float* b3 = (const float*)d_in[6];
    const float* y0 = (const float*)d_in[7];
    const float* L  = (const float*)d_in[8];
    float* out = (float*)d_out;  // [4096*100] x, then [4096] y

    char* ws = (char*)d_ws;
    size_t off = 0;
    float* dWpad = (float*)(ws + off);     off += (size_t)NSTEP * BATCH * DPAD * 4;
    _Float16* xin = (_Float16*)(ws + off); off += (size_t)NSTEP * BATCH * KU * 2;
    _Float16* WiT = (_Float16*)(ws + off); off += (size_t)NSTEP * BATCH * KU * 2;
    _Float16* W1h = (_Float16*)(ws + off); off += (size_t)512 * KU * 2;
    _Float16* W2h = (_Float16*)(ws + off); off += (size_t)512 * 512 * 2;
    _Float16* W3h = (_Float16*)(ws + off); off += (size_t)DPAD * 512 * 2;
    _Float16* Lh  = (_Float16*)(ws + off); off += (size_t)DPAD * KU * 2;
    float* zw = (float*)(ws + off);

    const int wtot = 512 * KU + 512 * 512 + DPAD * 512 + DPAD * KU;
    cvt_weights<<<(wtot + 255) / 256, 256, 0, stream>>>(W1, W2, W3, L,
                                                        W1h, W2h, W3h, Lh);
    transpose_wi<<<BATCH, 256, 0, stream>>>(W_indep, WiT);
    corr_gemm<<<(BATCH * NSTEP) / 64, 512, 0, stream>>>(WiT, Lh, dWpad);
    paths<<<BATCH, 128, 0, stream>>>(dWpad, xin, out);
    dim3 grid(BATCH / 64, NSTEP);
    mlp_fused<<<grid, 512, 0, stream>>>(xin, W1h, b1, W2h, b2, W3h, b3,
                                        dWpad, zw);
    finalize<<<16, 256, 0, stream>>>(zw, y0, out + (size_t)BATCH * 100);
}
